// MiniMoE_19748259627301
// MI455X (gfx1250) — matrix operand order repaired and checked by reading, not compiled, not run on hardware
//
#include <hip/hip_runtime.h>
#include <hip/hip_bf16.h>
#include <stdint.h>

// MiniMoE sparse dispatch for gfx1250 (MI455X), bf16 WMMA path, M=32 tiles,
// explicit double-buffered inner loop, padded-stride LDS (affine addressing).
// T=8192 tokens, D=1024, E=8 experts, K=2 experts/token.

#define T_TOK 8192
#define D_DIM 1024
#define N_EXP 8
#define TOPK  2
#define TK_SLOTS (T_TOK * TOPK)      // 16384 (token,slot) pairs
#define TILE_M 32                    // gathered rows per block
#define NTHREADS 512                 // 16 waves: 2 m-tiles x 8 n-groups(128 cols)
#define LDS_STRIDE 1032              // +8 bf16 pad: row stride 516 dwords == 4 mod 64
                                     // -> lane m reads bank-quads 4m..4m+3, conflict-free

typedef __bf16 bf16_t;
typedef __attribute__((ext_vector_type(16))) __bf16 v16bf;
typedef __attribute__((ext_vector_type(8)))  float  v8f;

union Frag32B { v16bf v; uint4 q[2]; };

__device__ __forceinline__ unsigned short f2bf(float f) {
    union { float f; unsigned u; } x; x.f = f;
    unsigned r = x.u + 0x7FFFu + ((x.u >> 16) & 1u);   // round-to-nearest-even
    return (unsigned short)(r >> 16);
}

// ---------------------------------------------------------------- zero out + counters
__global__ __launch_bounds__(256) void moe_zero_kernel(float* __restrict__ out,
                                                       int* __restrict__ cnt) {
    int i = blockIdx.x * 256 + threadIdx.x;
    const int n4 = (T_TOK * D_DIM) / 4;
    if (i < n4) {
        float4 z; z.x = 0.f; z.y = 0.f; z.z = 0.f; z.w = 0.f;
        ((float4*)out)[i] = z;
    }
    if (i < N_EXP) cnt[i] = 0;
}

// ---------------------------------------------------------------- f32 -> bf16 weights
__global__ __launch_bounds__(256) void moe_convert_kernel(const float* __restrict__ W1,
                                                          const float* __restrict__ W2,
                                                          unsigned short* __restrict__ W1b,
                                                          unsigned short* __restrict__ W2b) {
    size_t i4 = ((size_t)blockIdx.x * 256 + threadIdx.x) * 4;
    const size_t total = (size_t)N_EXP * D_DIM * D_DIM;  // 8.39M per tensor
    if (i4 >= total) return;
    float4 a = *(const float4*)(W1 + i4);
    float4 b = *(const float4*)(W2 + i4);
    uint2 pa, pb;
    pa.x = (unsigned)f2bf(a.x) | ((unsigned)f2bf(a.y) << 16);
    pa.y = (unsigned)f2bf(a.z) | ((unsigned)f2bf(a.w) << 16);
    pb.x = (unsigned)f2bf(b.x) | ((unsigned)f2bf(b.y) << 16);
    pb.y = (unsigned)f2bf(b.z) | ((unsigned)f2bf(b.w) << 16);
    *(uint2*)(W1b + i4) = pa;
    *(uint2*)(W2b + i4) = pb;
}

// ---------------------------------------------------------------- routing
__global__ __launch_bounds__(256) void moe_route_kernel(const int* __restrict__ idx,
                                                        int* __restrict__ cnt,
                                                        int* __restrict__ lists) {
    int i = blockIdx.x * 256 + threadIdx.x;
    if (i < TK_SLOTS) {
        int e = idx[i];
        int p = atomicAdd(&cnt[e], 1);
        lists[e * TK_SLOTS + p] = i;   // flat slot index; token = i/2, weight = wts[i]
    }
}

// Double-buffered GEMM inner loop (A from LDS, B streamed from bf16 weights).
// Final-iteration prefetch overreads are bounded and harmless: A reads land in
// LDS allocation slack (OOB LDS loads return 0 per ISA), B reads <=96B into
// adjacent workspace. All prefetched overread values are discarded.
#define GEMM_LOOP(Wp, cacc)                                                    \
  {                                                                            \
    const unsigned short* bbase = (Wp) + (size_t)(nbase + ml) * D_DIM + 8*hl;  \
    const unsigned short* abase = &a_lds[arow * LDS_STRIDE + 8 * hl];          \
    Frag32B a_cur, b_cur;                                                      \
    a_cur.q[0] = *(const uint4*)(abase);                                       \
    a_cur.q[1] = *(const uint4*)(abase + 16);                                  \
    b_cur.q[0] = *(const uint4*)(bbase);                                       \
    b_cur.q[1] = *(const uint4*)(bbase + 16);                                  \
    for (int k0 = 0; k0 < D_DIM; k0 += 32) {                                   \
      Frag32B a_nxt;                                                           \
      a_nxt.q[0] = *(const uint4*)(abase + k0 + 32);                           \
      a_nxt.q[1] = *(const uint4*)(abase + k0 + 48);                           \
      _Pragma("unroll")                                                        \
      for (int t = 0; t < 8; ++t) {                                            \
        Frag32B b_nxt;                                                         \
        const unsigned short* pn = (t < 7)                                     \
            ? bbase + (size_t)(t + 1) * (16 * D_DIM) + k0                      \
            : bbase + (k0 + 32);                                               \
        b_nxt.q[0] = *(const uint4*)pn;                                        \
        b_nxt.q[1] = *(const uint4*)(pn + 16);                                 \
        cacc[t] = __builtin_amdgcn_wmma_f32_16x16x32_bf16(                     \
                      false, a_cur.v, false, b_cur.v, (short)0, cacc[t],       \
                      false, false);                                           \
        b_cur = b_nxt;                                                         \
      }                                                                        \
      a_cur = a_nxt;                                                           \
    }                                                                          \
  }

// ---------------------------------------------------------------- fused expert tile
// Block: 512 threads = 16 waves. One block = one expert, 32 gathered rows.
// Wave w: m-tile mt = w>>3 (rows mt*16..mt*16+15), n-group ng = w&7 (cols ng*128..+127).
__global__ __launch_bounds__(NTHREADS) void moe_expert_kernel(
    const float* __restrict__ x,
    const float* __restrict__ wts,
    const unsigned short* __restrict__ W1b,
    const unsigned short* __restrict__ W2b,
    const int* __restrict__ cnt,
    const int* __restrict__ lists,
    float* __restrict__ out)
{
    const int e    = blockIdx.y;
    const int tile = blockIdx.x;
    const int n_e  = cnt[e];
    const int row0 = tile * TILE_M;
    if (row0 >= n_e) return;                       // uniform per block
    const int rows = (n_e - row0) < TILE_M ? (n_e - row0) : TILE_M;

    // x tile, then h tile; 32 x 1032 bf16 = 66,048 B (CDNA5 WGP has 320 KB LDS)
    __shared__ __align__(16) unsigned short a_lds[TILE_M * LDS_STRIDE];

    const int tid = threadIdx.x;

    // ---- gather 32 rows of x, f32 -> bf16, into padded-stride LDS
    {
        const int r   = tid >> 4;      // row 0..31
        const int seg = tid & 15;      // 64 floats per segment
        const bool valid = (r < rows);
        int tok = 0;
        if (valid) tok = lists[e * TK_SLOTS + row0 + r] >> 1;
        const float* src = x + (size_t)tok * D_DIM + seg * 64;
        unsigned short* dst = &a_lds[r * LDS_STRIDE + seg * 64];
        #pragma unroll
        for (int i = 0; i < 64; i += 4) {
            float4 v;
            if (valid) v = *(const float4*)(src + i);
            else { v.x = 0.f; v.y = 0.f; v.z = 0.f; v.w = 0.f; }
            uint2 p;
            p.x = (unsigned)f2bf(v.x) | ((unsigned)f2bf(v.y) << 16);
            p.y = (unsigned)f2bf(v.z) | ((unsigned)f2bf(v.w) << 16);
            *(uint2*)(dst + i) = p;
        }
    }
    __syncthreads();

    const int lane  = tid & 31;
    const int hl    = lane >> 4;              // lane half (0/1)
    const int ml    = lane & 15;              // m (A) or n (B/C) within 16
    const int wq    = tid >> 5;               // wave 0..15
    const int mt    = wq >> 3;                // m-tile 0/1
    const int nbase = (wq & 7) * 128;         // n-group base
    const int arow  = mt * 16 + ml;           // A-frag row for this lane

    const v8f vzero = {0.f, 0.f, 0.f, 0.f, 0.f, 0.f, 0.f, 0.f};

    // ================= GEMM1: h = relu(x @ W1^T) =================
    v8f c1[8];
    #pragma unroll
    for (int t = 0; t < 8; ++t) c1[t] = vzero;
    GEMM_LOOP(W1b + (size_t)e * D_DIM * D_DIM, c1);

    __syncthreads();   // everyone done reading x tile; reuse buffer for h
    #pragma unroll
    for (int t = 0; t < 8; ++t) {
        const int n = nbase + t * 16 + ml;
        #pragma unroll
        for (int r = 0; r < 8; ++r) {
            const int m = mt * 16 + r + 8 * hl;
            float v = c1[t][r];
            v = v > 0.f ? v : 0.f;                      // relu
            a_lds[m * LDS_STRIDE + n] = f2bf(v);
        }
    }
    __syncthreads();

    // ================= GEMM2: y = relu(h @ W2^T) =================
    v8f c2[8];
    #pragma unroll
    for (int t = 0; t < 8; ++t) c2[t] = vzero;
    GEMM_LOOP(W2b + (size_t)e * D_DIM * D_DIM, c2);

    // ---- epilogue: relu, weight-scale, scatter-add
    // (each out element receives exactly 2 commutative f32 adds -> deterministic)
    int   tok_r[8];
    float w_r[8];
    #pragma unroll
    for (int r = 0; r < 8; ++r) {
        const int m = mt * 16 + r + 8 * hl;
        if (m < rows) {
            int fi   = lists[e * TK_SLOTS + row0 + m];
            tok_r[r] = fi >> 1;
            w_r[r]   = wts[fi];
        } else {
            tok_r[r] = -1;
            w_r[r]   = 0.f;
        }
    }
    #pragma unroll
    for (int t = 0; t < 8; ++t) {
        const int n = nbase + t * 16 + ml;
        #pragma unroll
        for (int r = 0; r < 8; ++r) {
            if (tok_r[r] >= 0) {
                float v = c2[t][r];
                v = v > 0.f ? v : 0.f;
                atomicAdd(&out[(size_t)tok_r[r] * D_DIM + n], w_r[r] * v);
            }
        }
    }
}

// ---------------------------------------------------------------- launcher
extern "C" void kernel_launch(void* const* d_in, const int* in_sizes, int n_in,
                              void* d_out, int out_size, void* d_ws, size_t ws_size,
                              hipStream_t stream) {
    const float* x   = (const float*)d_in[0];
    const int*   fei = (const int*)  d_in[1];
    const float* few = (const float*)d_in[2];
    const float* W1  = (const float*)d_in[3];
    const float* W2  = (const float*)d_in[4];
    float* out = (float*)d_out;

    // workspace layout
    const size_t wbytes = (size_t)N_EXP * D_DIM * D_DIM * sizeof(unsigned short); // 16 MB
    char* ws = (char*)d_ws;
    unsigned short* W1b = (unsigned short*)ws;
    unsigned short* W2b = (unsigned short*)(ws + wbytes);
    int* cnt   = (int*)(ws + 2 * wbytes);
    int* lists = (int*)(ws + 2 * wbytes + 256);       // N_EXP * TK_SLOTS ints

    const int zb = (T_TOK * D_DIM / 4 + 255) / 256;
    moe_zero_kernel<<<zb, 256, 0, stream>>>(out, cnt);

    const int cb = (int)(((size_t)N_EXP * D_DIM * D_DIM / 4 + 255) / 256);
    moe_convert_kernel<<<cb, 256, 0, stream>>>(W1, W2, W1b, W2b);

    moe_route_kernel<<<(TK_SLOTS + 255) / 256, 256, 0, stream>>>(fei, cnt, lists);

    dim3 grid(TK_SLOTS / TILE_M, N_EXP);   // 512 tiles (worst case) x 8 experts; empties exit
    moe_expert_kernel<<<grid, NTHREADS, 0, stream>>>(x, few, W1b, W2b, cnt, lists, out);
}
